// SelfAttn2_77773267796568
// MI455X (gfx1250) — compile-verified, matrix operand order checked by reading, not run on hardware
//
#include <hip/hip_runtime.h>
#include <cstdint>
#include <cstddef>

// ---------------------------------------------------------------------------
// CDNA5 / gfx1250: all contractions on v_wmma_f32_16x16x32_f16 (f32 accum),
// f16 storage for the dominant (4,1024,16384) attention maps, TDM async
// tile loads for f16 B operands when the toolchain exposes the builtin.
// ---------------------------------------------------------------------------

typedef __attribute__((ext_vector_type(16))) _Float16 v16h;
typedef __attribute__((ext_vector_type(8)))  _Float16 v8h;
typedef __attribute__((ext_vector_type(8)))  float    v8f;
typedef unsigned int u32x4 __attribute__((ext_vector_type(4)));
typedef int          i32x8 __attribute__((ext_vector_type(8)));
typedef int          i32x4 __attribute__((ext_vector_type(4)));

union V16H { v16h v; v8h h[2]; };

#define BM 64
#define BN 128
#define BK 32
#define BKP 40   // padded LDS row (f16): 80B row stride, keeps 16B alignment

#if defined(__has_builtin)
#  if __has_builtin(__builtin_amdgcn_tensor_load_to_lds) && \
      __has_builtin(__builtin_amdgcn_s_wait_tensorcnt)
#    define ATT_TDM 1
#  endif
#endif
#ifndef ATT_TDM
#  define ATT_TDM 0
#endif

// Generic batched GEMM:  C = alpha * (op(A)*B + bias[m]) + beta * Cin
//   op(A) = A (MxK, lda) or A^T with A stored (KxM, lda) if TRANS_A.
//   B (KxN, ldb) row-major; C/Cin (MxN, ldc) row-major.
// Block = 256 threads = 8 wave32. Block tile 64x128, K-step 32.
// Wave tile 32x32: 2 A-frags x 2 B-frags -> 4 WMMAs per K-step.
template<bool TRANS_A, typename TA, typename TB, typename TC>
__global__ __launch_bounds__(256) void gemm_wmma(
    const TA* __restrict__ A, const TB* __restrict__ B,
    const float* __restrict__ bias, const float* __restrict__ Cin,
    const float* __restrict__ alpha_ptr, float alpha_c, float beta,
    TC* __restrict__ C,
    int K, int lda, int ldb, int ldc,
    long long sA, long long sB, long long sC, long long sCin)
{
    __shared__ _Float16 As[BM][BKP];   // [m][k]
    __shared__ _Float16 Bs[BN][BKP];   // [n][k] (fragment-ready, transposed)
#if ATT_TDM
    __shared__ _Float16 Braw[BK][BN];  // TDM landing pad, [k][n] as in memory
#endif

#if ATT_TDM
    constexpr bool useTdm = (sizeof(TB) == 2);
#else
    constexpr bool useTdm = false;
#endif

    const int bz = blockIdx.z;
    A += bz * sA;  B += bz * sB;  C += bz * sC;
    if (Cin) Cin += bz * sCin;

    const int m0 = blockIdx.y * BM;
    const int n0 = blockIdx.x * BN;
    const int t    = threadIdx.x;
    const int wave = t >> 5;
    const int lane = t & 31;
    const int wm = (wave & 1) * 32;     // wave's 32-row strip
    const int wn = (wave >> 1) * 32;    // wave's 32-col strip

    const int lrow  = lane & 15;
    const int khalf = (lane >> 4) << 3; // 16-bit fragment layout: 0 or 8

    v8f acc00 = {}, acc01 = {}, acc10 = {}, acc11 = {};

    for (int kt = 0; kt < K; kt += BK) {
        // ---- stage A tile (64x32 -> f16 LDS) ------------------------------
        {
            const int m  = t >> 2;            // 0..63
            const int kk = (t & 3) << 3;      // 0,8,16,24
            if (!TRANS_A) {
                const TA* src = A + (long long)(m0 + m) * lda + kt + kk;
                #pragma unroll
                for (int i = 0; i < 8; ++i) As[m][kk + i] = (_Float16)src[i];
            } else {
                #pragma unroll
                for (int i = 0; i < 8; ++i)
                    As[m][kk + i] = (_Float16)A[(long long)(kt + kk + i) * lda + (m0 + m)];
            }
        }
        // ---- stage B tile (32x128) ----------------------------------------
        if constexpr (useTdm) {
#if ATT_TDM
            // Tensor Data Mover: async 2D tile (tile_dim0=128 elems of 2B,
            // tile_dim1=32 rows, row stride = ldb) -> LDS, issued by wave 0.
            if (t < 32) {
                const unsigned long long ga =
                    (unsigned long long)(const void*)(B + (long long)kt * ldb + n0);
                const unsigned ldsBase = (unsigned)(unsigned long long)&Braw[0][0];
                u32x4 g0 = { 1u,                               // count=1 (valid D#)
                             ldsBase,                          // lds_addr [63:32]
                             (unsigned)ga,                     // global_addr lo
                             (unsigned)((ga >> 32) & 0x01FFFFFFu) | 0x80000000u }; // type=2
                i32x8 g1 = { (int)0x00010000,                  // data_size=1 (2 bytes)
                             (int)(((unsigned)ldb & 0xFFFFu) << 16),          // tensor_dim0 lo
                             (int)((((unsigned)ldb >> 16) & 0xFFFFu)
                                   | (((unsigned)K & 0xFFFFu) << 16)),        // dim0 hi | dim1 lo
                             (int)((((unsigned)K >> 16) & 0xFFFFu) | (128u << 16)), // dim1 hi | tile_dim0
                             32,                               // tile_dim1
                             ldb,                              // tensor_dim0_stride lo
                             0, 0 };
                i32x4 gz  = { 0, 0, 0, 0 };
                i32x8 gz8 = { 0, 0, 0, 0, 0, 0, 0, 0 };
                __builtin_amdgcn_tensor_load_to_lds(g0, g1, gz, gz, gz8, 0);
                __builtin_amdgcn_s_wait_tensorcnt(0);
            }
#endif
        } else {
            const int kk = t >> 3;            // 0..31
            const int nb = (t & 7) << 4;      // 0..112
            const TB* src = B + (long long)(kt + kk) * ldb + n0 + nb;
            #pragma unroll
            for (int i = 0; i < 16; ++i) Bs[nb + i][kk] = (_Float16)src[i];
            if (kt + BK < K)                  // global_prefetch_b8 for next tile
                __builtin_prefetch(B + (long long)(kt + BK + kk) * ldb + n0, 0, 1);
        }
        __syncthreads();
        if constexpr (useTdm) {
#if ATT_TDM
            // LDS transpose pass: [k][n] -> fragment-ready [n][k] (padded)
            const int kk = t >> 3;
            const int nb = (t & 7) << 4;
            #pragma unroll
            for (int i = 0; i < 16; ++i) Bs[nb + i][kk] = Braw[kk][nb + i];
#endif
            __syncthreads();
        }

        // ---- fragments (ds_load_b128) + 4 WMMAs ---------------------------
        V16H a0, a1, b0, b1;
        a0.h[0] = *(const v8h*)&As[wm + lrow][khalf];
        a0.h[1] = *(const v8h*)&As[wm + lrow][16 + khalf];
        a1.h[0] = *(const v8h*)&As[wm + 16 + lrow][khalf];
        a1.h[1] = *(const v8h*)&As[wm + 16 + lrow][16 + khalf];
        b0.h[0] = *(const v8h*)&Bs[wn + lrow][khalf];
        b0.h[1] = *(const v8h*)&Bs[wn + lrow][16 + khalf];
        b1.h[0] = *(const v8h*)&Bs[wn + 16 + lrow][khalf];
        b1.h[1] = *(const v8h*)&Bs[wn + 16 + lrow][16 + khalf];

        acc00 = __builtin_amdgcn_wmma_f32_16x16x32_f16(false, a0.v, false, b0.v, (short)0, acc00, false, false);
        acc01 = __builtin_amdgcn_wmma_f32_16x16x32_f16(false, a0.v, false, b1.v, (short)0, acc01, false, false);
        acc10 = __builtin_amdgcn_wmma_f32_16x16x32_f16(false, a1.v, false, b0.v, (short)0, acc10, false, false);
        acc11 = __builtin_amdgcn_wmma_f32_16x16x32_f16(false, a1.v, false, b1.v, (short)0, acc11, false, false);
        __syncthreads();
    }

    // ---- epilogue: C = alpha*(acc + bias) + beta*Cin -----------------------
    const float alpha = alpha_ptr ? *alpha_ptr : alpha_c;
    const int cn = n0 + wn + lrow;                 // lane's column (and +16)
    const int rb = m0 + wm + ((lane >> 4) << 3);   // first row of row-group 0
    #pragma unroll
    for (int g = 0; g < 2; ++g) {
        const v8f* aL = g ? &acc10 : &acc00;
        const v8f* aR = g ? &acc11 : &acc01;
        #pragma unroll
        for (int r = 0; r < 8; ++r) {
            const int m = rb + 16 * g + r;
            const float bv = bias ? bias[m] : 0.0f;
            float c0 = alpha * ((*aL)[r] + bv);
            float c1 = alpha * ((*aR)[r] + bv);
            if (Cin) {
                c0 += beta * Cin[(long long)m * ldc + cn];
                c1 += beta * Cin[(long long)m * ldc + cn + 16];
            }
            C[(long long)m * ldc + cn]      = (TC)c0;
            C[(long long)m * ldc + cn + 16] = (TC)c1;
        }
    }
}

// ---------------------------------------------------------------------------
// Column softmax over axis=0 of (R x Ncols), in place; f32 math, T storage.
// One wave32 per column; online max/sum + __shfl_xor reduction.
// ADDMODE: 0 none, 1 += I (r==col), 2 += E (r == pooled(col)).
// ---------------------------------------------------------------------------
template<int ADDMODE, typename T>
__global__ __launch_bounds__(256) void col_softmax(
    const T* __restrict__ in, T* __restrict__ out,
    int R, int Ncols, long long sIn, long long sOut)
{
    in  += blockIdx.z * sIn;
    out += blockIdx.z * sOut;
    const int wave = threadIdx.x >> 5, lane = threadIdx.x & 31;
    const int col = blockIdx.x * 8 + wave;
    if (col >= Ncols) return;

    float m = -3.0e38f, s = 0.0f;
    for (int r = lane; r < R; r += 32) {
        const float v = (float)in[(long long)r * Ncols + col];
        const float mn = fmaxf(m, v);
        s = s * __expf(m - mn) + __expf(v - mn);
        m = mn;
    }
    #pragma unroll
    for (int off = 16; off; off >>= 1) {
        const float mo = __shfl_xor(m, off, 32);
        const float so = __shfl_xor(s, off, 32);
        const float mn = fmaxf(m, mo);
        s = s * __expf(m - mn) + so * __expf(mo - mn);
        m = mn;
    }
    const float inv = 1.0f / s;

    int pi = -1;
    if (ADDMODE == 1) pi = col;
    if (ADDMODE == 2) { const int h = col >> 7, w = col & 127; pi = ((h >> 2) << 5) | (w >> 2); }

    for (int r = lane; r < R; r += 32) {
        float v = __expf((float)in[(long long)r * Ncols + col] - m) * inv;
        if (ADDMODE != 0 && r == pi) v += 1.0f;
        out[(long long)r * Ncols + col] = (T)v;
    }
}

// 4x4 average pool of qkv channels [chanBase, chanBase+64) -> (4, 32, 1024)
__global__ void pool4x4(const float* __restrict__ qkv, float* __restrict__ dst, int chanBase)
{
    const int idx = blockIdx.x * blockDim.x + threadIdx.x;
    if (idx >= 4 * 32 * 1024) return;
    const int i1 = idx & 1023, c1 = (idx >> 10) & 31, b1 = idx >> 15;
    const int b = b1 >> 1, g = b1 & 1;
    const float* src = qkv + ((long long)b * 384 + chanBase + g * 32 + c1) * 16384;
    const int h1 = i1 >> 5, w1 = i1 & 31;
    float sum = 0.0f;
    #pragma unroll
    for (int dy = 0; dy < 4; ++dy)
        #pragma unroll
        for (int dx = 0; dx < 4; ++dx)
            sum += src[(h1 * 4 + dy) * 128 + (w1 * 4 + dx)];
    dst[idx] = sum * 0.0625f;
}

__global__ void zero_n(float* p, int n) { int i = blockIdx.x*256 + threadIdx.x; if (i < n) p[i] = 0.0f; }

// per-batch inf-norm (max row sum) of 1024x1024 M (uint atomicMax on +floats)
__global__ void rowsum_max(const float* __restrict__ M, float* __restrict__ norms)
{
    const int row = blockIdx.x * 256 + threadIdx.x;
    const float* src = M + (long long)blockIdx.z * 1048576 + (long long)row * 1024;
    float s = 0.0f;
    for (int j = 0; j < 1024; ++j) s += src[j];
    atomicMax((unsigned*)&norms[blockIdx.z], __float_as_uint(s));
}

// X0 = M^T / (||M||_1 * ||M||_inf); ||M||_1 == 2 exactly (col-stochastic + I)
__global__ void transpose_scale(const float* __restrict__ in,
                                const float* __restrict__ norms,
                                float* __restrict__ out)
{
    const long long base = (long long)blockIdx.z * 1048576;
    const int idx = blockIdx.x * 256 + threadIdx.x;
    const int i = idx >> 10, j = idx & 1023;
    const float scale = 1.0f / (2.0f * norms[blockIdx.z]);
    out[base + (long long)i * 1024 + j] = in[base + (long long)j * 1024 + i] * scale;
}

// ---------------------------------------------------------------------------
template<bool TRA, typename TAe, typename TBe, typename TCe>
static inline void launch_gemm(const TAe* A, const TBe* B, const float* bias,
                               const float* Cin, const float* aP, float aC, float bt,
                               TCe* C, int M, int N, int K, int lda, int ldb, int ldc,
                               long long sA, long long sB, long long sC, long long sCin,
                               int batch, hipStream_t s)
{
    dim3 g(N / BN, M / BM, batch), b(256);
    hipLaunchKernelGGL((gemm_wmma<TRA, TAe, TBe, TCe>), g, b, 0, s,
                       A, B, bias, Cin, aP, aC, bt, C, K, lda, ldb, ldc, sA, sB, sC, sCin);
}

extern "C" void kernel_launch(void* const* d_in, const int* in_sizes, int n_in,
                              void* d_out, int out_size, void* d_ws, size_t ws_size,
                              hipStream_t stream)
{
    const float* x     = (const float*)d_in[0];   // (2,256,128,128)
    const float* qkv_w = (const float*)d_in[1];   // (384,256)
    const float* qkv_b = (const float*)d_in[2];   // (384)
    const float* out_w = (const float*)d_in[3];   // (256,256)
    const float* out_b = (const float*)d_in[4];   // (256)
    const float* gamma = (const float*)d_in[5];   // (1) device scalar
    float* out = (float*)d_out;                   // (2,256,128,128)

    // workspace carve-up
    float* f = (float*)d_ws;
    size_t o = 0;
    float* qkv = f + o; o += 12582912ULL;     // (2,384,16384) f32
    float* qb  = f + o; o += 131072ULL;       // (4,32,1024)   pooled q
    float* kb  = f + o; o += 131072ULL;       // (4,32,1024)   pooled k
    float* Mm  = f + o; o += 4194304ULL;      // (4,1024,1024) softmax(k^T q)+I
    float* Xa  = f + o; o += 4194304ULL;      // Newton ping
    float* Xb  = f + o; o += 4194304ULL;      // Newton pong
    float* Yb  = f + o; o += 4194304ULL;      // M*X
    _Float16* S1h = (_Float16*)(f + o); o += 33554432ULL;  // (4,1024,16384) f16
    _Float16* S2h = (_Float16*)(f + o); o += 33554432ULL;  // (4,1024,16384) f16
    float* Tb  = f + o; o += 524288ULL;       // (4,128,1024)
    float* Ob  = f + o; o += 8388608ULL;      // (4,128,16384) == (2,256,16384)
    float* nrm = f + o; o += 4ULL;            // per-batch ||M||_inf

    const dim3 blk(256);

    // 1) qkv = qkv_w @ x + qkv_b                       (batched over B=2)
    launch_gemm<false>(qkv_w, x, qkv_b, (const float*)nullptr, (const float*)nullptr,
                       1.f, 0.f, qkv, 384, 16384, 256, 256, 16384, 16384,
                       0LL, 4194304LL, 6291456LL, 0LL, 2, stream);

    // 2) pooled q, k
    hipLaunchKernelGGL(pool4x4, dim3(512), blk, 0, stream, qkv, qb, 0);
    hipLaunchKernelGGL(pool4x4, dim3(512), blk, 0, stream, qkv, kb, 64);

    // 3) M = softmax(k^T q) + I                        (4 head-groups)
    launch_gemm<true>(kb, qb, (const float*)nullptr, (const float*)nullptr,
                      (const float*)nullptr, 1.f, 0.f, Mm,
                      1024, 1024, 32, 1024, 1024, 1024,
                      32768LL, 32768LL, 1048576LL, 0LL, 4, stream);
    hipLaunchKernelGGL((col_softmax<1, float>), dim3(128, 1, 4), blk, 0, stream,
                       Mm, Mm, 1024, 1024, 1048576LL, 1048576LL);

    // 4) inv(M): Newton-Schulz, X0 = M^T/(2*||M||_inf), X <- 2X - X(MX)
    hipLaunchKernelGGL(zero_n, dim3(1), blk, 0, stream, nrm, 4);
    hipLaunchKernelGGL(rowsum_max, dim3(4, 1, 4), blk, 0, stream, Mm, nrm);
    hipLaunchKernelGGL(transpose_scale, dim3(4096, 1, 4), blk, 0, stream, Mm, nrm, Xa);
    float* Xc = Xa; float* Xn = Xb;
    for (int it = 0; it < 20; ++it) {
        launch_gemm<false>(Mm, Xc, (const float*)nullptr, (const float*)nullptr,
                           (const float*)nullptr, 1.f, 0.f, Yb,
                           1024, 1024, 1024, 1024, 1024, 1024,
                           1048576LL, 1048576LL, 1048576LL, 0LL, 4, stream);
        launch_gemm<false>(Xc, Yb, (const float*)nullptr, Xc,
                           (const float*)nullptr, -1.f, 2.f, Xn,
                           1024, 1024, 1024, 1024, 1024, 1024,
                           1048576LL, 1048576LL, 1048576LL, 1048576LL, 4, stream);
        float* tp = Xc; Xc = Xn; Xn = tp;
    }

    // 5) A_c = softmax(k^T Qf) + E  -> f16             (2 launches, group stride)
    for (int b = 0; b < 2; ++b)
        launch_gemm<true>(kb + (long long)b * 65536, qkv + (long long)b * 6291456,
                          (const float*)nullptr, (const float*)nullptr,
                          (const float*)nullptr, 1.f, 0.f,
                          S1h + (long long)b * 33554432,
                          1024, 16384, 32, 1024, 16384, 16384,
                          32768LL, 524288LL, 16777216LL, 0LL, 2, stream);
    hipLaunchKernelGGL((col_softmax<2, _Float16>), dim3(2048, 1, 4), blk, 0, stream,
                       S1h, S1h, 1024, 16384, 16777216LL, 16777216LL);

    // 6) S2 = inv(M) @ (A_c + E)   (f16 B via TDM path, f16 out)
    launch_gemm<false>(Xc, S1h, (const float*)nullptr, (const float*)nullptr,
                       (const float*)nullptr, 1.f, 0.f, S2h,
                       1024, 16384, 1024, 1024, 16384, 16384,
                       1048576LL, 16777216LL, 16777216LL, 0LL, 4, stream);

    // 7) A_s = softmax(Kf^T q) -> f16 (reuse S1)
    for (int b = 0; b < 2; ++b)
        launch_gemm<true>(qkv + ((long long)b * 384 + 64) * 16384,
                          qb + (long long)b * 65536,
                          (const float*)nullptr, (const float*)nullptr,
                          (const float*)nullptr, 1.f, 0.f,
                          S1h + (long long)b * 33554432,
                          16384, 1024, 32, 16384, 1024, 1024,
                          524288LL, 32768LL, 16777216LL, 0LL, 2, stream);
    hipLaunchKernelGGL((col_softmax<0, _Float16>), dim3(128, 1, 4), blk, 0, stream,
                       S1h, S1h, 16384, 1024, 16777216LL, 16777216LL);

    // 8) T = Vf @ A_s              (f16 B via TDM path)
    for (int b = 0; b < 2; ++b)
        launch_gemm<false>(qkv + ((long long)b * 384 + 128) * 16384,
                           S1h + (long long)b * 33554432,
                           (const float*)nullptr, (const float*)nullptr,
                           (const float*)nullptr, 1.f, 0.f,
                           Tb + (long long)b * 262144,
                           128, 1024, 16384, 16384, 1024, 1024,
                           2097152LL, 16777216LL, 131072LL, 0LL, 2, stream);

    // 9) O = T @ S2                (f16 B via TDM path)
    launch_gemm<false>(Tb, S2h, (const float*)nullptr, (const float*)nullptr,
                       (const float*)nullptr, 1.f, 0.f, Ob,
                       128, 16384, 1024, 1024, 16384, 16384,
                       131072LL, 16777216LL, 2097152LL, 0LL, 4, stream);

    // 10) out = gamma*(out_w @ O + out_b) + x          (mode==0 in harness)
    launch_gemm<false>(out_w, Ob, out_b, x, gamma, 1.f, 1.f, out,
                       256, 16384, 256, 256, 16384, 16384,
                       0LL, 4194304LL, 4194304LL, 4194304LL, 2, stream);
}
